// Head_21552145891400
// MI455X (gfx1250) — matrix operand order repaired and checked by reading, not compiled, not run on hardware
//
#include <hip/hip_runtime.h>
#include <hip/hip_bf16.h>

// ---------------------------------------------------------------------------
// Causal single-head attention, B=4, L=4096, E=768, D=64.
//   cvt_x: x f32 -> bf16 (bandwidth-bound pass)
//   cvt_w: Wq/Wk/Wv f32 [768,64] -> bf16 transposed [64,768]
//   qkv:   bf16 WMMA GEMMs, software-pipelined; Q pre-scaled; V transposed
//   attn:  flash attention; K/V blocks DMA'd into LDS by the Tensor Data
//          Mover (double-buffered, TENSORcnt), bf16 WMMA, online softmax
// ---------------------------------------------------------------------------

#define B_  4
#define L_  4096
#define E_  768
#define D_  64
#define M_  (B_ * L_)          // 16384 rows
#define LOG2E 1.44269504088896340736f

typedef __attribute__((ext_vector_type(16))) __bf16    v16bf;
typedef __attribute__((ext_vector_type(8)))  float     v8f;
typedef __attribute__((ext_vector_type(4)))  unsigned  u32x4;
typedef __attribute__((ext_vector_type(8)))  int       i32x8;
typedef __attribute__((ext_vector_type(4)))  int       i32x4;

union Frag {
    v16bf v;
    unsigned u[8];
};

__device__ __forceinline__ unsigned short f2bf(float f) {
    unsigned u = __builtin_bit_cast(unsigned, f);
    u += 0x7FFFu + ((u >> 16) & 1u);
    return (unsigned short)(u >> 16);
}

__device__ __forceinline__ unsigned pack2(float lo, float hi) {
    unsigned a = __builtin_bit_cast(unsigned, lo);
    unsigned b = __builtin_bit_cast(unsigned, hi);
    a += 0x7FFFu + ((a >> 16) & 1u);
    b += 0x7FFFu + ((b >> 16) & 1u);
    return (a >> 16) | (b & 0xFFFF0000u);
}

__device__ __forceinline__ v8f wmma_bf16(const Frag& a, const Frag& b, v8f c) {
    return __builtin_amdgcn_wmma_f32_16x16x32_bf16(
        false, a.v, false, b.v, (short)0, c, false, false);
}

// ---------------------------------------------------------------------------
// TDM: async DMA of a 2D bf16 tile (64 wide x 64 rows) into LDS, with 1 dword
// of padding appended per 32-dword (64-element) row => LDS row stride 66
// shorts, bank-conflict-free fragment reads.  D# per CDNA5 ISA sec.8.
// Toolchain here exposes the 6-arg builtin:
//   (u32x4 g0, i32x8 g1, i32x4 g2, i32x4 g3, i32x8 extra, i32 cpol)
// ---------------------------------------------------------------------------
#if __has_builtin(__builtin_amdgcn_tensor_load_to_lds)
#define HAVE_TDM 1
__device__ __forceinline__ void tdm_load_tile64(
    unsigned lds_byte_off, unsigned long long gaddr,
    unsigned stride0_elems)
{
    u32x4 g0;
    g0[0] = 1u;                                          // count=1, user mode
    g0[1] = lds_byte_off;                                // lds_addr
    g0[2] = (unsigned)(gaddr & 0xFFFFFFFFull);           // global_addr[31:0]
    g0[3] = (unsigned)((gaddr >> 32) & 0x01FFFFFFull)    // global_addr[56:32]
          | (2u << 30);                                  // type = 2 (image)
    i32x8 g1;
    g1[0] = (int)((1u << 16)       // data_size = 1 -> 2 bytes
                | (1u << 20)       // pad_enable
                | (4u << 22));     // pad_interval = 4 -> every 32 dwords
                                   // pad_amount  = 0 -> 1 dword
    g1[1] = (int)((64u & 0xFFFFu) << 16);                // tensor_dim0 = 64
    g1[2] = (int)((64u & 0xFFFFu) << 16);                // td0 hi=0 | tensor_dim1 = 64
    g1[3] = (int)((64u & 0xFFFFu) << 16);                // td1 hi=0 | tile_dim0 = 64
    g1[4] = (int)(64u);                                  // tile_dim1 = 64, tile_dim2 = 0
    g1[5] = (int)stride0_elems;                          // tensor_dim0_stride[31:0]
    g1[6] = 0;                                           // stride0 hi | stride1 lo
    g1[7] = 0;                                           // stride1 hi
    i32x4 z4 = {0, 0, 0, 0};
    i32x8 z8 = {0, 0, 0, 0, 0, 0, 0, 0};
    __builtin_amdgcn_tensor_load_to_lds(g0, g1, z4, z4, z8, 0);
}
#define TDM_WAIT(n) __builtin_amdgcn_s_wait_tensorcnt((unsigned short)(n))
#else
#define HAVE_TDM 0
// Fallback: cooperative synchronous copy with the same padded layout.
__device__ __forceinline__ void fallback_load_tile64(
    unsigned short* dst, const unsigned short* src, unsigned stride0_elems,
    int lane)
{
    for (int i = lane; i < 512; i += 32) {               // 512 x uint4 = 8 KB
        int row = i >> 3, seg = i & 7;
        uint4 v = *(const uint4*)(src + (size_t)row * stride0_elems + seg * 8);
        *(uint4*)(dst + row * 66 + seg * 8) = v;
    }
}
#define TDM_WAIT(n) ((void)0)
#endif

// ---------------------------------------------------------------------------
// Kernel 0a: x f32 -> bf16.  grid = 6144, block = 256
// ---------------------------------------------------------------------------
__global__ __launch_bounds__(256) void cvt_x_kernel(
    const float* __restrict__ x, unsigned short* __restrict__ xb)
{
    const size_t i = ((size_t)blockIdx.x * 256 + threadIdx.x) * 8;
    float4 f0 = *(const float4*)(x + i);
    float4 f1 = *(const float4*)(x + i + 4);
    uint4 o;
    o.x = pack2(f0.x, f0.y);
    o.y = pack2(f0.z, f0.w);
    o.z = pack2(f1.x, f1.y);
    o.w = pack2(f1.z, f1.w);
    *(uint4*)(xb + i) = o;
}

// ---------------------------------------------------------------------------
// Kernel 0b: W -> Wt [3][64][768] bf16 transposed.  grid = 288, block = 256
// ---------------------------------------------------------------------------
__global__ __launch_bounds__(256) void cvt_w_kernel(
    const float* __restrict__ Wq, const float* __restrict__ Wk,
    const float* __restrict__ Wv, unsigned* __restrict__ Wt32)
{
    const int idx = blockIdx.x * 256 + threadIdx.x;
    const int mat = idx / (D_ * E_ / 2);
    const int rem = idx % (D_ * E_ / 2);
    const int n   = rem / (E_ / 2);
    const int kp  = rem % (E_ / 2);
    const float* W = (mat == 0) ? Wq : ((mat == 1) ? Wk : Wv);
    const int k = kp * 2;
    Wt32[idx] = pack2(W[(size_t)k * D_ + n], W[(size_t)(k + 1) * D_ + n]);
}

// ---------------------------------------------------------------------------
// Kernel 1: QKV projection, two-stage software pipeline.
// grid = 1024 (16-row tiles), block = 96 (3 waves: Q / K / V^T)
// ---------------------------------------------------------------------------
__global__ __launch_bounds__(96) void qkv_proj_kernel(
    const unsigned* __restrict__ xb32,       // bf16 pairs of x [M, E/2]
    const unsigned* __restrict__ Wt32,       // bf16 pairs [3][64][E/2]
    const float* __restrict__ bq, const float* __restrict__ bk,
    const float* __restrict__ bv,
    unsigned short* __restrict__ Qb,
    unsigned short* __restrict__ Kb,
    unsigned short* __restrict__ Vt)
{
    const int wave  = threadIdx.x >> 5;
    const int lane  = threadIdx.x & 31;
    const int lhalf = lane >> 4;
    const int l15   = lane & 15;
    const int m0    = blockIdx.x * 16;

    const unsigned* Wm = Wt32 + (size_t)wave * (D_ * E_ / 2);
    const float* bias  = (wave == 0) ? bq : ((wave == 1) ? bk : bv);

    v8f acc[4];
    #pragma unroll
    for (int t = 0; t < 4; ++t)
        #pragma unroll
        for (int r = 0; r < 8; ++r) acc[t][r] = 0.0f;

    const unsigned* xrow = xb32 + (size_t)(m0 + l15) * (E_ / 2);

    auto loadStage = [&](int ks, Frag& a, Frag (&b)[4]) {
        const int k0h = ks * 16;
        #pragma unroll
        for (int j = 0; j < 8; ++j) {
            int kkh = k0h + lhalf * 8 + j;
            a.u[j] = xrow[kkh];
        }
        #pragma unroll
        for (int t = 0; t < 4; ++t) {
            const unsigned* wrow =
                Wm + (size_t)(t * 16 + l15) * (E_ / 2) + k0h + lhalf * 8;
            #pragma unroll
            for (int j = 0; j < 8; ++j) b[t].u[j] = wrow[j];
        }
    };

    Frag aP, bP[4], aN, bN[4];
    loadStage(0, aP, bP);
    for (int ks = 0; ks < E_ / 32; ks += 2) {
        loadStage(ks + 1, aN, bN);           // prefetch odd stage
        #pragma unroll
        for (int t = 0; t < 4; ++t) acc[t] = wmma_bf16(aP, bP[t], acc[t]);
        if (ks + 2 < E_ / 32) loadStage(ks + 2, aP, bP);   // prefetch even
        #pragma unroll
        for (int t = 0; t < 4; ++t) acc[t] = wmma_bf16(aN, bN[t], acc[t]);
    }

    #pragma unroll
    for (int t = 0; t < 4; ++t) {
        #pragma unroll
        for (int r = 0; r < 8; ++r) {
            const int m   = lhalf * 8 + r;
            const int n   = t * 16 + l15;
            const int row = m0 + m;
            float v = acc[t][r] + bias[n];
            if (wave == 0) {
                v *= 0.125f;                         // 1/sqrt(D) folded into Q
                Qb[(size_t)row * D_ + n] = f2bf(v);
            } else if (wave == 1) {
                Kb[(size_t)row * D_ + n] = f2bf(v);
            } else {
                const int bb  = row >> 12;
                const int pos = row & 4095;
                Vt[((size_t)bb * D_ + n) * L_ + pos] = f2bf(v);
            }
        }
    }
}

// ---------------------------------------------------------------------------
// Kernel 2: flash attention, single-wave workgroups (block = 32, grid = 1024).
// K/V 64-key blocks are DMA'd into LDS by the TDM (double-buffered, padded
// rows -> stride 66 shorts).  Compute reads fragments from LDS (ds_load).
// ---------------------------------------------------------------------------
__global__ __launch_bounds__(32) void attn_kernel(
    const unsigned short* __restrict__ Qb,
    const unsigned short* __restrict__ Kb,
    const unsigned short* __restrict__ Vt,
    float* __restrict__ out)
{
    __shared__ unsigned short Kbuf[2][64 * 66];   // [key_local][d], padded
    __shared__ unsigned short Vbuf[2][64 * 66];   // [d][key_local], padded
    __shared__ unsigned short Pst[16 * 66];       // P staging, padded

    const int lane  = threadIdx.x;
    const int lhalf = lane >> 4;
    const int l15   = lane & 15;

    const int tile = blockIdx.x;                 // 1024 tiles
    const int bb   = tile >> 8;
    const int q0   = (tile & 255) << 4;

    const unsigned*       Q32 = (const unsigned*)(Qb + (size_t)bb * L_ * D_);
    const unsigned short* Kba = Kb + (size_t)bb * L_ * D_;
    const unsigned short* Vba = Vt + (size_t)bb * D_ * L_;
#if HAVE_TDM
    const unsigned long long Kga = (unsigned long long)(size_t)Kba;
    const unsigned long long Vga = (unsigned long long)(size_t)Vba;
    const unsigned koff0 = (unsigned)(size_t)&Kbuf[0][0];
    const unsigned koff1 = (unsigned)(size_t)&Kbuf[1][0];
    const unsigned voff0 = (unsigned)(size_t)&Vbuf[0][0];
    const unsigned voff1 = (unsigned)(size_t)&Vbuf[1][0];
#endif

    // resident Q fragments: d-slices 0..31 / 32..63
    Frag aq[2];
    #pragma unroll
    for (int s = 0; s < 2; ++s) {
        #pragma unroll
        for (int j = 0; j < 8; ++j) {
            int kk = 32 * s + lhalf * 16 + 2 * j;
            aq[s].u[j] = Q32[(q0 + l15) * (D_ / 2) + (kk >> 1)];
        }
    }

    v8f o[4];
    float m_i[8], l_i[8];
    #pragma unroll
    for (int t = 0; t < 4; ++t)
        #pragma unroll
        for (int r = 0; r < 8; ++r) o[t][r] = 0.0f;
    #pragma unroll
    for (int r = 0; r < 8; ++r) { m_i[r] = -3.0e38f; l_i[r] = 0.0f; }

    const int jmax = q0 + 15;

    // prologue: DMA first K/V block into buffer 0
#if HAVE_TDM
    tdm_load_tile64(koff0, Kga, D_);             // K rows are stride-64
    tdm_load_tile64(voff0, Vga, L_);             // V^T rows are stride-4096
#else
    fallback_load_tile64(&Kbuf[0][0], Kba, D_, lane);
    fallback_load_tile64(&Vbuf[0][0], Vba, L_, lane);
#endif

    int cur = 0;
    for (int j0 = 0; j0 <= jmax; j0 += 64) {
        // issue DMA for the next block into the other buffer, then wait for
        // the current block (in-order TENSORcnt: <=2 leaves next in flight)
        if (j0 + 64 <= jmax) {
#if HAVE_TDM
            tdm_load_tile64(cur ? koff0 : koff1,
                            Kga + (unsigned long long)(j0 + 64) * D_ * 2, D_);
            tdm_load_tile64(cur ? voff0 : voff1,
                            Vga + (unsigned long long)(j0 + 64) * 2, L_);
            TDM_WAIT(2);
#else
            fallback_load_tile64(&Kbuf[cur ^ 1][0],
                                 Kba + (size_t)(j0 + 64) * D_, D_, lane);
            fallback_load_tile64(&Vbuf[cur ^ 1][0],
                                 Vba + (j0 + 64), L_, lane);
#endif
        } else {
            TDM_WAIT(0);
        }
        asm volatile("" ::: "memory");
        const unsigned short* Kc = &Kbuf[cur][0];
        const unsigned short* Vc = &Vbuf[cur][0];

        // ---- S = Q . K^T : four 16-key tiles, fragments from LDS ----
        v8f S[4];
        #pragma unroll
        for (int c = 0; c < 4; ++c)
            #pragma unroll
            for (int r = 0; r < 8; ++r) S[c][r] = 0.0f;
        #pragma unroll
        for (int c = 0; c < 4; ++c) {
            const unsigned short* krow = Kc + (16 * c + l15) * 66;
            #pragma unroll
            for (int s = 0; s < 2; ++s) {
                Frag bk;
                #pragma unroll
                for (int j = 0; j < 8; ++j) {
                    int dd = 32 * s + lhalf * 16 + 2 * j;
                    bk.u[j] = *(const unsigned*)(krow + dd);
                }
                S[c] = wmma_bf16(aq[s], bk, S[c]);
            }
        }

        // ---- causal mask (diagonal block only) + online softmax ----
        const bool diag = (j0 + 63 > q0);
        float p[4][8], sc[8];
        #pragma unroll
        for (int r = 0; r < 8; ++r) {
            const int qg = q0 + lhalf * 8 + r;
            float s0 = S[0][r], s1 = S[1][r], s2 = S[2][r], s3 = S[3][r];
            if (diag) {
                if (j0      + l15 > qg) s0 = -3.0e38f;
                if (j0 + 16 + l15 > qg) s1 = -3.0e38f;
                if (j0 + 32 + l15 > qg) s2 = -3.0e38f;
                if (j0 + 48 + l15 > qg) s3 = -3.0e38f;
            }
            float rm = fmaxf(fmaxf(s0, s1), fmaxf(s2, s3));
            #pragma unroll
            for (int mk = 1; mk <= 8; mk <<= 1)
                rm = fmaxf(rm, __shfl_xor(rm, mk, 32));
            const float mn = fmaxf(m_i[r], rm);
            sc[r]   = __builtin_amdgcn_exp2f((m_i[r] - mn) * LOG2E);
            p[0][r] = __builtin_amdgcn_exp2f((s0 - mn) * LOG2E);
            p[1][r] = __builtin_amdgcn_exp2f((s1 - mn) * LOG2E);
            p[2][r] = __builtin_amdgcn_exp2f((s2 - mn) * LOG2E);
            p[3][r] = __builtin_amdgcn_exp2f((s3 - mn) * LOG2E);
            float rs = (p[0][r] + p[1][r]) + (p[2][r] + p[3][r]);
            #pragma unroll
            for (int mk = 1; mk <= 8; mk <<= 1)
                rs += __shfl_xor(rs, mk, 32);
            l_i[r] = l_i[r] * sc[r] + rs;
            m_i[r] = mn;
        }

        #pragma unroll
        for (int t = 0; t < 4; ++t)
            #pragma unroll
            for (int r = 0; r < 8; ++r) o[t][r] *= sc[r];

        // ---- stage P (C layout) -> LDS -> reload as two A fragments ----
        #pragma unroll
        for (int r = 0; r < 8; ++r) {
            const int m = lhalf * 8 + r;
            #pragma unroll
            for (int c = 0; c < 4; ++c)
                Pst[m * 66 + 16 * c + l15] = f2bf(p[c][r]);
        }
        asm volatile("" ::: "memory");
        Frag ap[2];
        #pragma unroll
        for (int h = 0; h < 2; ++h) {
            #pragma unroll
            for (int j = 0; j < 8; ++j) {
                int kk = 32 * h + lhalf * 16 + 2 * j;
                ap[h].u[j] = *(const unsigned*)(&Pst[0] + l15 * 66 + kk);
            }
        }
        asm volatile("" ::: "memory");

        // ---- O += P . V : fragments from LDS ----
        #pragma unroll
        for (int t = 0; t < 4; ++t) {
            const unsigned short* vrow = Vc + (t * 16 + l15) * 66;
            #pragma unroll
            for (int h = 0; h < 2; ++h) {
                Frag bv_;
                #pragma unroll
                for (int j = 0; j < 8; ++j) {
                    int kl = 32 * h + lhalf * 16 + 2 * j;
                    bv_.u[j] = *(const unsigned*)(vrow + kl);
                }
                o[t] = wmma_bf16(ap[h], bv_, o[t]);
            }
        }
        cur ^= 1;
    }

    // ---- epilogue: normalize, store f32 ----
    float* orow = out + ((size_t)bb * L_ + q0) * D_;
    #pragma unroll
    for (int t = 0; t < 4; ++t) {
        #pragma unroll
        for (int r = 0; r < 8; ++r) {
            const int m = lhalf * 8 + r;
            const int n = t * 16 + l15;
            orow[(size_t)m * D_ + n] = o[t][r] / l_i[r];
        }
    }
}

// ---------------------------------------------------------------------------
extern "C" void kernel_launch(void* const* d_in, const int* in_sizes, int n_in,
                              void* d_out, int out_size, void* d_ws, size_t ws_size,
                              hipStream_t stream) {
    // setup_inputs order: x, causal_mask, pad_mask, Wk, bk, Wq, bq, Wv, bv
    const float* x  = (const float*)d_in[0];
    const float* Wk = (const float*)d_in[3];
    const float* bk = (const float*)d_in[4];
    const float* Wq = (const float*)d_in[5];
    const float* bq = (const float*)d_in[6];
    const float* Wv = (const float*)d_in[7];
    const float* bv = (const float*)d_in[8];

    unsigned short* xb   = (unsigned short*)d_ws;                 // [M,E] bf16
    unsigned*       Wt32 = (unsigned*)(xb + (size_t)M_ * E_);     // [3][64][E/2]
    unsigned short* Qb   = (unsigned short*)(Wt32 + 3 * D_ * E_ / 2);
    unsigned short* Kb   = Qb + (size_t)M_ * D_;
    unsigned short* Vt   = Kb + (size_t)M_ * D_;

    cvt_x_kernel<<<(M_ * E_) / (256 * 8), 256, 0, stream>>>(x, xb);
    cvt_w_kernel<<<(3 * D_ * E_ / 2) / 256, 256, 0, stream>>>(Wq, Wk, Wv, Wt32);
    qkv_proj_kernel<<<M_ / 16, 96, 0, stream>>>((const unsigned*)xb, Wt32,
                                                bq, bk, bv, Qb, Kb, Vt);
    attn_kernel<<<B_ * L_ / 16, 32, 0, stream>>>(Qb, Kb, Vt, (float*)d_out);
}
